// GraphPolicy_11398843203995
// MI455X (gfx1250) — matrix operand order repaired and checked by reading, not compiled, not run on hardware
//
#include <hip/hip_runtime.h>
#include <hip/hip_bf16.h>
#include <math.h>

// ---------------------------------------------------------------------------
// GraphPolicy forward for MI455X (gfx1250).
// Memory/atomic-bound workload (~50 GFLOP dense vs ~GBs of edge scatter);
// dense projections use v_wmma_f32_16x16x32_bf16 (wave32, 16x16 tiles) with
// LDS-staged bf16 fragments (ds_load_b128 path); segment ops use float
// atomics to HBM/L2.
// ---------------------------------------------------------------------------

#define EMBD 64
#define HCD  128
#define NLAY 3
#define NGR  512   // num_graphs fixed by setup_inputs(); device value read where it matters

typedef __bf16 bf16_t;
typedef __attribute__((ext_vector_type(16))) bf16_t v16bf;
typedef __attribute__((ext_vector_type(8)))  bf16_t v8bf;
typedef __attribute__((ext_vector_type(8)))  float  v8f;

__device__ __forceinline__ float leaky01(float v) { return v > 0.f ? v : 0.01f * v; }

__device__ __forceinline__ void atomicMaxF(float* addr, float val) {
  if (val >= 0.f) atomicMax((int*)addr, __float_as_int(val));
  else            atomicMin((unsigned int*)addr, __float_as_uint(val));
}

// ------------------------- WMMA GEMM --------------------------------------
// C[M,N] = act(A[M,K] @ W[K,N] + bias), K % 32 == 0.
// Block = 128 threads (4 waves); block tile = 64 rows x 16 cols.
// A (64x32) and B^T (16x32) staged in LDS as bf16; fragments via ds b128 loads.
__global__ void __launch_bounds__(128)
gemm_wmma_kernel(const float* __restrict__ A,
                 const float* __restrict__ W,
                 const float* __restrict__ bias,
                 float* __restrict__ C,
                 int M, int N, int K, int act)
{
  __shared__ __align__(16) bf16_t sA[64 * 32];
  __shared__ __align__(16) bf16_t sBT[16 * 32];

  const int tid  = threadIdx.x;   // 0..127
  const int lane = tid & 31;
  const int wv   = tid >> 5;      // wave id 0..3 -> 16-row slice of the tile
  const int half = lane >> 4;     // 0: lanes 0-15, 1: lanes 16-31
  const int l16  = lane & 15;
  const long row0 = (long)blockIdx.x * 64;
  const int  col0 = blockIdx.y * 16;

  v8f acc = {};
  for (int kk = 0; kk < K; kk += 32) {
    __syncthreads();
    // ---- stage A tile: 64 rows x 32 k, coalesced float4 loads -> bf16 LDS
#pragma unroll
    for (int q = 0; q < 4; ++q) {
      int s   = q * 128 + tid;    // 512 float4 slots
      int row = s >> 3;
      int kq  = (s & 7) * 4;
      float4 v = make_float4(0.f, 0.f, 0.f, 0.f);
      long grow = row0 + row;
      if (grow < M) v = *reinterpret_cast<const float4*>(&A[grow * K + kk + kq]);
      sA[row * 32 + kq + 0] = (bf16_t)v.x;
      sA[row * 32 + kq + 1] = (bf16_t)v.y;
      sA[row * 32 + kq + 2] = (bf16_t)v.z;
      sA[row * 32 + kq + 3] = (bf16_t)v.w;
    }
    // ---- stage B tile transposed: sBT[n][k], 32x16 elements
#pragma unroll
    for (int q = 0; q < 4; ++q) {
      int s  = q * 128 + tid;     // 512 elements
      int k  = s >> 4;
      int nn = s & 15;
      float v = 0.f;
      if (col0 + nn < N) v = W[(long)(kk + k) * N + col0 + nn];
      sBT[nn * 32 + k] = (bf16_t)v;
    }
    __syncthreads();
    // ---- fragments (ISA 7.12.2 layouts), contiguous 16B LDS loads
    int arow = wv * 16 + l16;
    const v8bf a0 = *reinterpret_cast<const v8bf*>(&sA[arow * 32 + 8 * half]);
    const v8bf a1 = *reinterpret_cast<const v8bf*>(&sA[arow * 32 + 16 + 8 * half]);
    const v8bf b0 = *reinterpret_cast<const v8bf*>(&sBT[l16 * 32 + 8 * half]);
    const v8bf b1 = *reinterpret_cast<const v8bf*>(&sBT[l16 * 32 + 16 + 8 * half]);
    v16bf fa = __builtin_shufflevector(a0, a1, 0,1,2,3,4,5,6,7,8,9,10,11,12,13,14,15);
    v16bf fb = __builtin_shufflevector(b0, b1, 0,1,2,3,4,5,6,7,8,9,10,11,12,13,14,15);
    acc = __builtin_amdgcn_wmma_f32_16x16x32_bf16(false, fa, false, fb,
                                                  (short)0, acc, false, false);
  }
  int col = col0 + l16;
  if (col < N) {
    float bv = bias ? bias[col] : 0.f;
#pragma unroll
    for (int d = 0; d < 8; ++d) {
      long row = row0 + wv * 16 + d + 8 * half;  // C/D: VGPR d -> M=d (+8 for upper lanes)
      if (row < M) {
        float v = acc[d] + bv;
        if (act == 1) v = leaky01(v);
        C[row * N + col] = v;
      }
    }
  }
}

// ------------------------- elementwise / scatter kernels -------------------
__device__ __forceinline__ long gtid() {
  return blockIdx.x * (long)blockDim.x + threadIdx.x;
}

__global__ void k_fill(float* p, float v, long n) {
  long i = gtid(); if (i < n) p[i] = v;
}

__global__ void k_embed(const int* nt, const int* nst, const int* fo,
                        const float* ent, const float* ens, const float* evirt,
                        float* x, int n, int m) {
  long t = gtid(); if (t >= (long)m * EMBD) return;
  int row = (int)(t >> 6), f = (int)(t & 63);
  if (row < n) {
    float v = ent[nt[row] * EMBD + f] + ens[nst[row] * EMBD + f];
    int p = fo[row];
    float pe = 0.f;
    if (p != -1) {
      int j = f & 31;
      float fr = powf(10000.f, -(float)(2 * j) / 64.f);
      float o = (float)p * fr;
      pe = (f < 32) ? sinf(o) : cosf(o);
    }
    x[t] = v + pe;
  } else {
    x[t] = evirt[f];
  }
}

__global__ void k_topology(const int* ei, const int* batch, int* src, int* dst,
                           int E0, int n, int E, long ET) {
  long i = gtid(); if (i >= ET) return;
  int s, d;
  if (i < E0)            { s = ei[i]; d = ei[E0 + i]; }
  else if (i < E0 + n)   { int u = (int)(i - E0);     s = u;            d = batch[u] + n; }
  else if (i < E)        { int u = (int)(i - E0 - n); s = batch[u] + n; d = u; }
  else                   { int u = (int)(i - E);      s = u;            d = u; }
  src[i] = s; dst[i] = d;
}

__global__ void k_ea_fill(const int* etype, const float* embE, float* ea,
                          int E0, long E) {
  long t = gtid(); if (t >= E * EMBD) return;
  long i = t >> 6; int f = (int)(t & 63);
  ea[t] = (i < E0) ? embE[etype[i] * EMBD + f] : (f == 0 ? 1.f : 0.f);
}

__global__ void k_loop_cnt(const int* dst, float* cnt, long E) {
  long i = gtid(); if (i < E) atomicAdd(&cnt[dst[i]], 1.f);
}

__global__ void k_loop_sum(const int* dst, const float* ea, float* lsum, long E) {
  long t = gtid(); if (t >= E * EMBD) return;
  long i = t >> 6; int f = (int)(t & 63);
  atomicAdd(&lsum[(long)dst[i] * EMBD + f], ea[t]);
}

__global__ void k_loop_write(const float* lsum, const float* lcnt, float* ea,
                             long E, int m) {
  long t = gtid(); if (t >= (long)m * EMBD) return;
  long row = t >> 6;
  ea[(E + row) * EMBD + (t & 63)] = lsum[t] / fmaxf(lcnt[row], 1.f);
}

__global__ void k_count_nodes(const int* batch, float* cntg, int n) {
  long i = gtid(); if (i < n) atomicAdd(&cntg[batch[i]], 1.f);
}

// ---- graph norm (mean/var over all features of a graph) ----
__global__ void k_gn_sum(const float* x, const int* batch, float* gsum, int n, int m) {
  long row = gtid(); if (row >= m) return;
  int gid = (row < n) ? batch[row] : (int)(row - n);
  float s = 0.f;
  for (int f = 0; f < EMBD; ++f) s += x[row * EMBD + f];
  atomicAdd(&gsum[gid], s);
}

__global__ void k_gn_var(const float* x, const int* batch, const float* gsum,
                         const float* cntg, float* gvar, int n, int m) {
  long row = gtid(); if (row >= m) return;
  int gid = (row < n) ? batch[row] : (int)(row - n);
  float c64 = fmaxf((cntg[gid] + 1.f) * 64.f, 1.f);
  float mean = gsum[gid] / c64;
  float s = 0.f;
  for (int f = 0; f < EMBD; ++f) { float d = x[row * EMBD + f] - mean; s += d * d; }
  atomicAdd(&gvar[gid], s);
}

__global__ void k_gn_norm(const float* x, const int* batch, const float* gsum,
                          const float* gvar, const float* cntg, float* out,
                          int n, int m) {
  long t = gtid(); if (t >= (long)m * EMBD) return;
  long row = t >> 6;
  int gid = (row < n) ? batch[row] : (int)(row - n);
  float c64 = fmaxf((cntg[gid] + 1.f) * 64.f, 1.f);
  float mean = gsum[gid] / c64;
  float rstd = rsqrtf(gvar[gid] / c64 + 1e-5f);
  out[t] = (x[t] - mean) * rstd;
}

__global__ void k_msg_agg(const float* hn, const float* ea, const int* src,
                          const int* dst, float* agg, long ET) {
  long t = gtid(); if (t >= ET * EMBD) return;
  long e = t >> 6; int f = (int)(t & 63);
  float v = fmaxf(hn[(long)src[e] * EMBD + f] + ea[t], 0.f) + 1e-7f;
  atomicAdd(&agg[(long)dst[e] * EMBD + f], v);
}

__global__ void k_add(const float* a, const float* b, float* c, long n) {
  long i = gtid(); if (i < n) c[i] = a[i] + b[i];
}

__global__ void k_concat128(const float* hn, const float* gen, float* xc, int m) {
  long t = gtid(); if (t >= (long)m * HCD) return;
  long row = t >> 7; int f = (int)(t & 127);
  xc[t] = (f < EMBD) ? hn[row * EMBD + f] : gen[row * EMBD + f - EMBD];
}

__global__ void k_alpha(const float* __restrict__ q, const float* __restrict__ k,
                        const float* __restrict__ ee, const int* __restrict__ src,
                        const int* __restrict__ dst, float* alpha, float* amax,
                        long ET) {
  long t = gtid(); if (t >= ET * 2) return;
  long e = t >> 1; int h = (int)(t & 1);
  const float* qd = q  + (long)dst[e] * HCD + h * EMBD;
  const float* ks = k  + (long)src[e] * HCD + h * EMBD;
  const float* ep = ee + e * HCD + h * EMBD;
  float s = 0.f;
#pragma unroll 8
  for (int j = 0; j < EMBD; ++j) s += qd[j] * (ks[j] + ep[j]);
  s *= 0.125f;  // 1/sqrt(64)
  alpha[t] = s;
  atomicMaxF(&amax[(long)dst[e] * 2 + h], s);
}

__global__ void k_amax_fix(float* amax, long n) {
  long i = gtid(); if (i >= n) return;
  float v = amax[i];
  if (!(v == v && v < INFINITY && v > -INFINITY)) amax[i] = 0.f;
}

__global__ void k_expa(const float* alpha, const float* amax, const int* dst,
                       float* expa, float* denom, long ET) {
  long t = gtid(); if (t >= ET * 2) return;
  long e = t >> 1; int h = (int)(t & 1);
  float ex = expf(alpha[t] - amax[(long)dst[e] * 2 + h]);
  expa[t] = ex;
  atomicAdd(&denom[(long)dst[e] * 2 + h], ex);
}

__global__ void k_attn(const float* vv, const float* ee, const float* expa,
                       const float* denom, const int* src, const int* dst,
                       float* attn, long ET) {
  long t = gtid(); if (t >= ET * HCD) return;
  long e = t >> 7; int f = (int)(t & 127); int h = f >> 6;
  long d = dst[e];
  float w = expa[e * 2 + h] / denom[d * 2 + h];
  atomicAdd(&attn[d * HCD + f], w * (vv[(long)src[e] * HCD + f] + ee[t]));
}

__global__ void k_glob_sum(const float* h, const int* batch, float* gsum2, int n) {
  long t = gtid(); if (t >= (long)n * EMBD) return;
  long row = t >> 6; int f = (int)(t & 63);
  atomicAdd(&gsum2[(long)batch[row] * EMBD + f], h[t]);
}

__global__ void k_glob(const float* gsum2, const float* cntg, const float* h,
                       float* glob, int n, int g) {
  long t = gtid(); if (t >= (long)g * EMBD) return;
  long row = t >> 6; int f = (int)(t & 63);
  glob[t] = gsum2[t] / fmaxf(cntg[row], 1.f) + h[((long)n + row) * EMBD + f];
}

__global__ void k_ne(const float* h, const int* nei, float* nee, long NE) {
  long t = gtid(); if (t >= NE * EMBD) return;
  long i = t >> 6; int f = (int)(t & 63);
  nee[t] = h[(long)nei[i] * EMBD + f] + h[(long)nei[NE + i] * EMBD + f];
}

__global__ void k_head_hidden(const float* glob, const float* hw1, const float* hb1,
                              const int* isp, const int* jsp, float* hid) {
  int t = (int)gtid(); if (t >= NGR * EMBD) return;
  int row = t >> 6, j = t & 63;
  int is = *isp, js = *jsp;
  int hd = row < is ? 0 : (row < js ? 1 : 2);
  const float* w = hw1 + (long)hd * EMBD * EMBD;
  float s = hb1[hd * EMBD + j];
  for (int kx = 0; kx < EMBD; ++kx) s += glob[row * EMBD + kx] * w[kx * EMBD + j];
  hid[t] = leaky01(s);
}

__global__ void k_head_out(const float* hid, const float* wi, const float* bi,
                           const float* wn, const float* bn2, const float* we1,
                           const float* be1, const int* isp, const int* jsp,
                           const int* ngp, float* out) {
  int t = (int)gtid(); if (t >= NGR * 81) return;
  int row = t / 81, j = t % 81;
  int is = *isp, js = *jsp, g = *ngp;
  if (row >= g) return;
  if (row < is) {
    if (j < 20) {
      float s = bi[j];
      for (int kx = 0; kx < EMBD; ++kx) s += hid[row * EMBD + kx] * wi[kx * 20 + j];
      out[row * 20 + j] = s;
    }
  } else if (row < js) {
    float s = bn2[j];
    for (int kx = 0; kx < EMBD; ++kx) s += hid[row * EMBD + kx] * wn[kx * 81 + j];
    out[(long)is * 20 + (long)(row - is) * 81 + j] = s;
  } else {
    if (j == 0) {
      float s = be1[0];
      for (int kx = 0; kx < EMBD; ++kx) s += hid[row * EMBD + kx] * we1[kx];
      out[(long)is * 20 + (long)(js - is) * 81 + (row - js)] = s;
    }
  }
}

__global__ void k_ne_out(const float* nehid, const float* w2, const float* b2,
                         const int* isp, const int* jsp, const int* ngp,
                         float* out, long NE) {
  long t = gtid(); if (t >= NE * 4) return;
  long i = t >> 2; int j = (int)(t & 3);
  int is = *isp, js = *jsp, g = *ngp;
  long base = (long)is * 20 + (long)(js - is) * 81 + (g - js);
  float s = b2[j];
  for (int kx = 0; kx < EMBD; ++kx) s += nehid[i * EMBD + kx] * w2[kx * 4 + j];
  out[base + i * 4 + j] = s;
}

// ---------------------------------------------------------------------------
extern "C" void kernel_launch(void* const* d_in, const int* in_sizes, int n_in,
                              void* d_out, int out_size, void* d_ws, size_t ws_size,
                              hipStream_t stream) {
  (void)n_in; (void)out_size; (void)ws_size;
  const int*   node_type = (const int*)d_in[0];
  const int*   node_st   = (const int*)d_in[1];
  const int*   frontier  = (const int*)d_in[2];
  const int*   edge_type = (const int*)d_in[3];
  const int*   edge_idx  = (const int*)d_in[4];
  const int*   batch     = (const int*)d_in[5];
  const int*   non_edge  = (const int*)d_in[6];
  const int*   d_isplit  = (const int*)d_in[7];
  const int*   d_jsplit  = (const int*)d_in[8];
  const int*   d_numg    = (const int*)d_in[9];
  const float* emb_nt    = (const float*)d_in[10];
  const float* emb_ns    = (const float*)d_in[11];
  const float* emb_et    = (const float*)d_in[12];
  const float* emb_v     = (const float*)d_in[13];
  const float* gen_w     = (const float*)d_in[14];
  const float* gen_b     = (const float*)d_in[15];
  const float* trans_w   = (const float*)d_in[16];
  const float* trans_b   = (const float*)d_in[17];
  const float* we_w      = (const float*)d_in[18];
  const float* lin_w     = (const float*)d_in[19];
  const float* lin_b     = (const float*)d_in[20];
  const float* ff_w1     = (const float*)d_in[21];
  const float* ff_b1     = (const float*)d_in[22];
  const float* ff_w2     = (const float*)d_in[23];
  const float* ff_b2     = (const float*)d_in[24];
  const float* head_w1   = (const float*)d_in[25];
  const float* head_b1   = (const float*)d_in[26];
  const float* init_w2   = (const float*)d_in[27];
  const float* init_b2   = (const float*)d_in[28];
  const float* nlv_w2    = (const float*)d_in[29];
  const float* nlv_b2    = (const float*)d_in[30];
  const float* e1_w2     = (const float*)d_in[31];
  const float* e1_b2     = (const float*)d_in[32];
  const float* e2_w2     = (const float*)d_in[33];
  const float* e2_b2     = (const float*)d_in[34];

  const int  n  = in_sizes[0];
  const int  E0 = in_sizes[4] / 2;
  const long NE = in_sizes[6] / 2;
  const int  g  = NGR;
  const int  m  = n + g;
  const int  E  = E0 + 2 * n;
  const long ET = (long)E + m;
  float* out = (float*)d_out;

  // ---- workspace bump allocator ----
  char* wp = (char*)d_ws;
  auto allocF = [&](long nf) { float* r = (float*)wp; wp += ((nf * sizeof(float) + 255) & ~(size_t)255); return r; };
  auto allocI = [&](long ni) { int*   r = (int*)wp;   wp += ((ni * sizeof(int)   + 255) & ~(size_t)255); return r; };

  int*   srcI  = allocI(ET);
  int*   dstI  = allocI(ET);
  float* hbuf  = allocF((long)m * EMBD);
  float* ea    = allocF(ET * EMBD);
  float* hn    = allocF((long)m * EMBD);
  float* agg   = allocF((long)m * EMBD);
  float* tmp64 = allocF((long)m * EMBD);
  float* gen   = allocF((long)m * EMBD);
  float* xc    = allocF((long)m * HCD);
  float* qb    = allocF((long)m * HCD);
  float* kb    = allocF((long)m * HCD);
  float* vb    = allocF((long)m * HCD);
  float* t4    = allocF((long)m * HCD);
  float* attn  = allocF((long)m * HCD);
  float* ee    = allocF(ET * HCD);
  float* alpha = allocF(ET * 2);
  float* expa  = allocF(ET * 2);
  float* amax  = allocF((long)m * 2);
  float* denom = allocF((long)m * 2);
  float* gsum  = allocF(g);
  float* gvar  = allocF(g);
  float* cntg  = allocF(g);
  float* gsum2 = allocF((long)g * EMBD);
  float* lsum  = allocF((long)m * EMBD);
  float* lcnt  = allocF(m);
  float* l_h   = allocF((long)m * EMBD);
  float* gn2   = allocF((long)m * EMBD);
  float* ff1   = allocF((long)m * 4 * EMBD);
  float* ff2   = allocF((long)m * EMBD);
  float* glob  = allocF((long)g * EMBD);
  float* hidg  = allocF((long)g * EMBD);
  float* nee   = allocF(NE * EMBD);
  float* nehid = allocF(NE * EMBD);

  auto zero = [&](void* p, long nf) { hipMemsetAsync(p, 0, nf * sizeof(float), stream); };
#define L1D(kern, total, ...) do { long _t = (total); int _b = (int)((_t + 255) / 256); \
    kern<<<dim3(_b), dim3(256), 0, stream>>>(__VA_ARGS__); } while (0)
  auto gemm = [&](const float* A, const float* W, const float* bias, float* C,
                  long M, int N, int K, int act) {
    dim3 blk(128);
    dim3 grd((unsigned)((M + 63) / 64), (unsigned)((N + 15) / 16));
    gemm_wmma_kernel<<<grd, blk, 0, stream>>>(A, W, bias, C, (int)M, N, K, act);
  };
  auto graphnorm = [&](const float* x, float* o) {
    zero(gsum, g); zero(gvar, g);
    L1D(k_gn_sum,  m, x, batch, gsum, n, m);
    L1D(k_gn_var,  m, x, batch, gsum, cntg, gvar, n, m);
    L1D(k_gn_norm, (long)m * EMBD, x, batch, gsum, gvar, cntg, o, n, m);
  };

  // ---- setup: embeddings, topology, edge attrs, loop attrs, counts ----
  zero(cntg, g); zero(lcnt, m); zero(lsum, (long)m * EMBD); zero(gsum2, (long)g * EMBD);
  L1D(k_embed, (long)m * EMBD, node_type, node_st, frontier, emb_nt, emb_ns, emb_v, hbuf, n, m);
  L1D(k_topology, ET, edge_idx, batch, srcI, dstI, E0, n, E, ET);
  L1D(k_ea_fill, (long)E * EMBD, edge_type, emb_et, ea, E0, (long)E);
  L1D(k_loop_cnt, (long)E, dstI, lcnt, (long)E);
  L1D(k_loop_sum, (long)E * EMBD, dstI, ea, lsum, (long)E);
  L1D(k_loop_write, (long)m * EMBD, lsum, lcnt, ea, (long)E, m);
  L1D(k_count_nodes, n, batch, cntg, n);

  // ---- transformer layers ----
  for (int l = 0; l < NLAY; ++l) {
    graphnorm(hbuf, hn);
    zero(agg, (long)m * EMBD);
    L1D(k_msg_agg, ET * EMBD, hn, ea, srcI, dstI, agg, ET);
    L1D(k_add, (long)m * EMBD, agg, hn, tmp64, (long)m * EMBD);
    gemm(tmp64, gen_w + (long)l * EMBD * EMBD, gen_b + l * EMBD, gen, m, EMBD, EMBD, 0);
    L1D(k_concat128, (long)m * HCD, hn, gen, xc, m);
    const long twsz = (long)HCD * HCD;
    gemm(xc, trans_w + ((long)l * 4 + 0) * twsz, trans_b + (l * 4 + 0) * HCD, qb, m, HCD, HCD, 0);
    gemm(xc, trans_w + ((long)l * 4 + 1) * twsz, trans_b + (l * 4 + 1) * HCD, kb, m, HCD, HCD, 0);
    gemm(xc, trans_w + ((long)l * 4 + 2) * twsz, trans_b + (l * 4 + 2) * HCD, vb, m, HCD, HCD, 0);
    gemm(xc, trans_w + ((long)l * 4 + 3) * twsz, trans_b + (l * 4 + 3) * HCD, t4, m, HCD, HCD, 0);
    gemm(ea, we_w + (long)l * EMBD * HCD, nullptr, ee, ET, HCD, EMBD, 0);
    L1D(k_fill, (long)m * 2, amax, -INFINITY, (long)m * 2);
    L1D(k_alpha, ET * 2, qb, kb, ee, srcI, dstI, alpha, amax, ET);
    L1D(k_amax_fix, (long)m * 2, amax, (long)m * 2);
    zero(denom, (long)m * 2);
    L1D(k_expa, ET * 2, alpha, amax, dstI, expa, denom, ET);
    zero(attn, (long)m * HCD);
    L1D(k_attn, ET * HCD, vb, ee, expa, denom, srcI, dstI, attn, ET);
    L1D(k_add, (long)m * HCD, attn, t4, attn, (long)m * HCD);
    gemm(attn, lin_w + (long)l * HCD * EMBD, lin_b + l * EMBD, l_h, m, EMBD, HCD, 0);
    graphnorm(l_h, gn2);
    gemm(gn2, ff_w1 + (long)l * EMBD * 4 * EMBD, ff_b1 + l * 4 * EMBD, ff1, m, 4 * EMBD, EMBD, 1);
    gemm(ff1, ff_w2 + (long)l * 4 * EMBD * EMBD, ff_b2 + l * EMBD, ff2, m, EMBD, 4 * EMBD, 0);
    L1D(k_add, (long)m * EMBD, hbuf, ff2, hbuf, (long)m * EMBD);
  }

  // ---- readout ----
  L1D(k_glob_sum, (long)n * EMBD, hbuf, batch, gsum2, n);
  L1D(k_glob, (long)g * EMBD, gsum2, cntg, hbuf, glob, n, g);
  L1D(k_ne, NE * EMBD, hbuf, non_edge, nee, NE);

  L1D(k_head_hidden, (long)NGR * EMBD, glob, head_w1, head_b1, d_isplit, d_jsplit, hidg);
  L1D(k_head_out, (long)NGR * 81, hidg, init_w2, init_b2, nlv_w2, nlv_b2,
      e1_w2, e1_b2, d_isplit, d_jsplit, d_numg, out);

  gemm(nee, head_w1 + 3L * EMBD * EMBD, head_b1 + 3 * EMBD, nehid, NE, EMBD, EMBD, 1);
  L1D(k_ne_out, NE * 4, nehid, e2_w2, e2_b2, d_isplit, d_jsplit, d_numg, out, NE);
#undef L1D
}
